// Router_3736621547980
// MI455X (gfx1250) — matrix operand order repaired and checked by reading, not compiled, not run on hardware
//
#include <hip/hip_runtime.h>

// ---------------------------------------------------------------------------
// MoE router for MI455X (gfx1250, wave32, WMMA, async-to-LDS).
//
// Roofline: x is 256 MB f32 -> ~11 us at 23.3 TB/s HBM; intensity 32 FLOP/B
// needs ~745 TF/s to stay memory bound -> bf16 WMMA (16x16x32), not f32
// (16x16x4). W is converted once to bf16 (512 KB, L2-resident); per K-chunk
// it is staged into LDS with GLOBAL_LOAD_ASYNC_TO_LDS_B128 (double-buffered,
// ASYNCcnt), so the WMMA loop never waits on vmem for B. The A (x) stream is
// software-pipelined one iteration ahead + prefetched.
// ---------------------------------------------------------------------------

#define HIDDEN        4096
#define EXPERTS       64
#define TOPK          8
#define T_TOTAL       16384
#define TOK_PER_BLOCK 128            // 8 waves * 16 tokens
#define KC            256            // K-chunk staged in LDS
#define NCHUNK        (HIDDEN / KC)  // 16
#define ROWB          (KC * 2 + 16)  // 528 B padded LDS row stride (16B aligned)
#define BUFSZ         (EXPERTS * ROWB)

typedef __bf16 bf16;
typedef __attribute__((ext_vector_type(16))) __bf16 v16bf;
typedef __attribute__((ext_vector_type(8)))  __bf16 v8bf;
typedef __attribute__((ext_vector_type(8)))  float  v8f;

// --- Kernel 1: W f32 -> bf16 (64*4096 = 262144 elements) -------------------
__global__ void convert_w_kernel(const float* __restrict__ W,
                                 bf16* __restrict__ Wb) {
    int i = (blockIdx.x * blockDim.x + threadIdx.x) * 4;
#pragma unroll
    for (int j = 0; j < 4; ++j) Wb[i + j] = (bf16)W[i + j];
}

__device__ __forceinline__ v16bf cvt16(float4 f0, float4 f1, float4 f2, float4 f3) {
    v16bf a;
    a[0]=(bf16)f0.x;  a[1]=(bf16)f0.y;  a[2]=(bf16)f0.z;  a[3]=(bf16)f0.w;
    a[4]=(bf16)f1.x;  a[5]=(bf16)f1.y;  a[6]=(bf16)f1.z;  a[7]=(bf16)f1.w;
    a[8]=(bf16)f2.x;  a[9]=(bf16)f2.y;  a[10]=(bf16)f2.z; a[11]=(bf16)f2.w;
    a[12]=(bf16)f3.x; a[13]=(bf16)f3.y; a[14]=(bf16)f3.z; a[15]=(bf16)f3.w;
    return a;
}

// --- Kernel 2: GEMM (bf16 WMMA, async-LDS B) + bias + softmax-top8 ---------
__global__ __launch_bounds__(256)
void router_kernel(const float* __restrict__ x,
                   const bf16*  __restrict__ Wb,
                   const float* __restrict__ bias,
                   float* __restrict__ out) {
    __shared__ __attribute__((aligned(16))) char smW[2 * BUFSZ];  // ~66 KB
    __shared__ float sm[TOK_PER_BLOCK][EXPERTS + 1];              // ~33 KB

    const int tid  = threadIdx.x;
    const int wave = tid >> 5;
    const int lane = tid & 31;
    const int half = lane >> 4;        // 0: lanes 0-15, 1: lanes 16-31
    const int l16  = lane & 15;
    const int koffA = half ? 8 : 0;    // A frag: K {0..7,16..23} / {8..15,24..31}
    const int koffB = half ? 8 : 0;    // B frag: K {0..7,16..23} / {8..15,24..31}

    const int tokenBase = blockIdx.x * TOK_PER_BLOCK + wave * 16;
    const float* xr = x + (size_t)(tokenBase + l16) * HIDDEN;

    // --- async stage of one W K-chunk into LDS: wave w owns rows w*8..w*8+7,
    //     one b128 per lane covers a full 512 B row ----------------------------
    auto issue_wchunk = [&](int c, char* dstbase) {
        const int kbase = c * KC;
#pragma unroll
        for (int r = 0; r < 8; ++r) {
            const int e = wave * 8 + r;
            const char* g = (const char*)Wb + (size_t)e * (HIDDEN * 2)
                          + (size_t)kbase * 2 + lane * 16;
            unsigned ldsoff = (unsigned)(uintptr_t)(dstbase + e * ROWB + lane * 16);
            asm volatile("global_load_async_to_lds_b128 %0, %1, off"
                         :: "v"(ldsoff), "v"(g) : "memory");
        }
    };

    v8f c0 = {}, c1 = {}, c2 = {}, c3 = {};

    // prologue: A registers for k = 0, first W chunk in flight
    float4 fa0 = *(const float4*)(xr + koffA);
    float4 fa1 = *(const float4*)(xr + koffA + 4);
    float4 fa2 = *(const float4*)(xr + 16 + koffA);
    float4 fa3 = *(const float4*)(xr + 16 + koffA + 4);
    issue_wchunk(0, smW);

    for (int c = 0; c < NCHUNK; ++c) {
        if (c + 1 < NCHUNK) {
            issue_wchunk(c + 1, smW + ((c + 1) & 1) * BUFSZ);
            asm volatile("s_wait_asynccnt 0x8" ::: "memory");  // chunk c landed
        } else {
            asm volatile("s_wait_asynccnt 0x0" ::: "memory");
        }
        __syncthreads();                                       // publish to all waves

        const char* bufp = smW + (c & 1) * BUFSZ;
#pragma unroll
        for (int i = 0; i < KC / 32; ++i) {
            const int kg = c * KC + i * 32;
            const int kn = (kg + 32) & (HIDDEN - 1);           // wrap avoids OOB tail

            // issue next iteration's A loads (pipelined one step ahead)
            float4 fb0 = *(const float4*)(xr + kn + koffA);
            float4 fb1 = *(const float4*)(xr + kn + koffA + 4);
            float4 fb2 = *(const float4*)(xr + kn + 16 + koffA);
            float4 fb3 = *(const float4*)(xr + kn + 16 + koffA + 4);
            __builtin_prefetch(xr + kg + 1024, 0, 1);          // global_prefetch_b8

            v16bf a = cvt16(fa0, fa1, fa2, fa3);

            // B fragments from LDS (ds_load_b128 x2 per tile)
            const int kb = i * 32 + koffB;
            const char* r0 = bufp + ( 0 + l16) * ROWB + kb * 2;
            const char* r1 = bufp + (16 + l16) * ROWB + kb * 2;
            const char* r2 = bufp + (32 + l16) * ROWB + kb * 2;
            const char* r3 = bufp + (48 + l16) * ROWB + kb * 2;
            v8bf lo, hi;
            lo = *(const v8bf*)r0;  hi = *(const v8bf*)(r0 + 32);
            v16bf b0 = __builtin_shufflevector(lo, hi, 0,1,2,3,4,5,6,7,8,9,10,11,12,13,14,15);
            lo = *(const v8bf*)r1;  hi = *(const v8bf*)(r1 + 32);
            v16bf b1 = __builtin_shufflevector(lo, hi, 0,1,2,3,4,5,6,7,8,9,10,11,12,13,14,15);
            lo = *(const v8bf*)r2;  hi = *(const v8bf*)(r2 + 32);
            v16bf b2 = __builtin_shufflevector(lo, hi, 0,1,2,3,4,5,6,7,8,9,10,11,12,13,14,15);
            lo = *(const v8bf*)r3;  hi = *(const v8bf*)(r3 + 32);
            v16bf b3 = __builtin_shufflevector(lo, hi, 0,1,2,3,4,5,6,7,8,9,10,11,12,13,14,15);

            c0 = __builtin_amdgcn_wmma_f32_16x16x32_bf16(false, a, false, b0, (short)0, c0, false, false);
            c1 = __builtin_amdgcn_wmma_f32_16x16x32_bf16(false, a, false, b1, (short)0, c1, false, false);
            c2 = __builtin_amdgcn_wmma_f32_16x16x32_bf16(false, a, false, b2, (short)0, c2, false, false);
            c3 = __builtin_amdgcn_wmma_f32_16x16x32_bf16(false, a, false, b3, (short)0, c3, false, false);

            fa0 = fb0; fa1 = fb1; fa2 = fb2; fa3 = fb3;
        }
        __syncthreads();   // all waves done reading buf (c&1) before it's refilled
    }

    // ---- Scatter logits (+bias) to LDS.  C layout: reg r, lanes 0-15 -> M=r,
    //      lanes 16-31 -> M=r+8; column N = lane%16. ----
#pragma unroll
    for (int r = 0; r < 8; ++r) {
        int tl = wave * 16 + r + half * 8;
        sm[tl][ 0 + l16] = c0[r] + bias[ 0 + l16];
        sm[tl][16 + l16] = c1[r] + bias[16 + l16];
        sm[tl][32 + l16] = c2[r] + bias[32 + l16];
        sm[tl][48 + l16] = c3[r] + bias[48 + l16];
    }
    __syncthreads();

    // ---- Per-token top-8 + renormalized softmax weights.
    //      softmax denominator cancels: w_k = exp(l_k - max) / sum_top8. ----
    if (tid < TOK_PER_BLOCK) {
        float* rowp = sm[tid];
        const int token = blockIdx.x * TOK_PER_BLOCK + tid;

        float vals[TOPK];
        int   idxs[TOPK];
#pragma unroll
        for (int k = 0; k < TOPK; ++k) {
            float best = -3.402823466e38f;
            int bi = 0;
            for (int j = 0; j < EXPERTS; ++j) {
                float v = rowp[j];
                if (v > best) { best = v; bi = j; }   // strict > : lowest-index tie-break
            }
            vals[k] = best; idxs[k] = bi;
            rowp[bi] = -3.402823466e38f;
        }

        const float m = vals[0];
        float e[TOPK];
        float sum = 0.f;
#pragma unroll
        for (int k = 0; k < TOPK; ++k) { e[k] = __expf(vals[k] - m); sum += e[k]; }
        const float inv = 1.0f / sum;

        float* wout = out + (size_t)token * TOPK;
        float* iout = out + (size_t)T_TOTAL * TOPK + (size_t)token * TOPK;
#pragma unroll
        for (int k = 0; k < TOPK; ++k) {
            wout[k] = e[k] * inv;
            iout[k] = (float)idxs[k];
        }
    }
}

// ---------------------------------------------------------------------------
extern "C" void kernel_launch(void* const* d_in, const int* in_sizes, int n_in,
                              void* d_out, int out_size, void* d_ws, size_t ws_size,
                              hipStream_t stream) {
    const float* x    = (const float*)d_in[0];
    const float* W    = (const float*)d_in[1];
    const float* bias = (const float*)d_in[2];
    float* out = (float*)d_out;
    bf16*  Wb  = (bf16*)d_ws;                 // 64*4096 bf16 = 512 KB scratch

    convert_w_kernel<<<dim3(256), dim3(256), 0, stream>>>(W, Wb);
    router_kernel<<<dim3(T_TOTAL / TOK_PER_BLOCK), dim3(256), 0, stream>>>(x, Wb, bias, out);
}
